// PerformerAttention_79860621902513
// MI455X (gfx1250) — matrix operand order repaired and checked by reading, not compiled, not run on hardware
//
#include <hip/hip_runtime.h>
#include <math.h>
#include <stdint.h>

// Problem constants (match reference)
constexpr int Bc = 4;
constexpr int Lc = 8192;
constexpr int Dc = 512;
constexpr int Mc = 256;
constexpr float EPSc = 1e-6f;
constexpr float SCALEc = 0.21022410381342865f; // 512^(-1/4)
constexpr float INV_SQRT_M = 0.0625f;          // 1/sqrt(256)

typedef __attribute__((ext_vector_type(16))) _Float16 v16h;
typedef __attribute__((ext_vector_type(8)))  _Float16 v8h;
typedef __attribute__((ext_vector_type(8)))  float    v8f;
typedef __attribute__((ext_vector_type(4)))  unsigned int u32x4;
typedef __attribute__((ext_vector_type(4)))  int i32x4;
typedef __attribute__((ext_vector_type(8)))  int i32x8;

// ---------------- WMMA fragment helpers (16x16x32 f16 -> f32) ----------------
// A (16x32 f16): lanes 0-15 -> M=lane, K {0..7,16..23}; lanes 16-31 -> K {8..15,24..31}
__device__ __forceinline__ v16h load_a_frag(const _Float16* __restrict__ Arow, int lda, int kk) {
    int lane = threadIdx.x & 31;
    int m = lane & 15;
    int kg = (lane >> 4) << 3; // 0 or 8
    const _Float16* p = Arow + (size_t)m * lda + kk + kg;
    v8h lo = *(const v8h*)p;          // K = kg .. kg+7
    v8h hi = *(const v8h*)(p + 16);   // K = 16+kg .. 16+kg+7
    v16h f;
#pragma unroll
    for (int i = 0; i < 8; ++i) { f[i] = lo[i]; f[8 + i] = hi[i]; }
    return f;
}

// B (32x16 f16) fragment from an LDS tile (row-major 32 x 256): N = lane%16, K {0..7,16..23} (lanes 0-15) / {8..15,24..31} (lanes 16-31)
__device__ __forceinline__ v16h lds_b_frag(const _Float16* base /* tile + n0 */) {
    int lane = threadIdx.x & 31;
    int n = lane & 15;
    int kg = (lane >> 4) << 3;
    v16h f;
#pragma unroll
    for (int i = 0; i < 8; ++i) { f[i] = base[(kg + i) * 256 + n]; f[8 + i] = base[(16 + kg + i) * 256 + n]; }
    return f;
}

// C/D (16x16 f32): VGPR i: lanes 0-15 -> M=i, N=lane; lanes 16-31 -> M=8+i, N=lane-16
__device__ __forceinline__ void store_c(float* __restrict__ C, int ldc, v8f c) {
    int lane = threadIdx.x & 31;
    int n = lane & 15;
    int mb = (lane >> 4) << 3;
#pragma unroll
    for (int i = 0; i < 8; ++i) C[(size_t)(mb + i) * ldc + n] = c[i];
}

#define WMMA_F16(a, b, c) \
    __builtin_amdgcn_wmma_f32_16x16x32_f16(false, (a), false, (b), (short)0, (c), false, false)

// ---------------- Tensor Data Mover staging (with portable fallback) ----------------
#define HAVE_TDM (__has_builtin(__builtin_amdgcn_tensor_load_to_lds) && \
                  __has_builtin(__builtin_amdgcn_s_wait_tensorcnt))

#if HAVE_TDM
// 2D tile load: tileRows x tileCols f16 elements, row stride strideElems, into LDS (contiguous).
// D# layout per CDNA5 ISA ch.8: group0 = {count, lds_addr, global_addr, type=2},
// group1 = {data_size=2B, tensor_dim0/1, tile_dim0/1, tensor_dim0_stride}. 2D -> groups 2/3 zero.
__device__ __forceinline__ void tdm_load_2d(void* ldsDst, const void* gsrc,
                                            unsigned tileCols, unsigned tileRows,
                                            unsigned long long strideElems) {
    unsigned lds_off = (unsigned)(uintptr_t)ldsDst;
    unsigned long long ga = (unsigned long long)(uintptr_t)gsrc;
    u32x4 g0;
    g0.x = 1u;                                                  // count=1, user descriptor
    g0.y = lds_off;                                             // lds_addr (bytes)
    g0.z = (unsigned)(ga & 0xffffffffu);                        // global_addr[31:0]
    g0.w = (unsigned)((ga >> 32) & 0x01ffffffu) | 0x80000000u;  // global_addr[56:32] | type=2
    unsigned long long td0 = tileCols;   // tensor_dim0
    unsigned long long td1 = tileRows;   // tensor_dim1
    unsigned long long s0  = strideElems;
    i32x8 g1;
    g1[0] = (int)(1u << 16);                                           // data_size=1 (2 bytes)
    g1[1] = (int)((td0 & 0xffffu) << 16);                              // tensor_dim0[15:0]
    g1[2] = (int)(((td0 >> 16) & 0xffffu) | ((td1 & 0xffffu) << 16));  // td0[31:16] | td1[15:0]
    g1[3] = (int)(((td1 >> 16) & 0xffffu) | ((tileCols & 0xffffu) << 16)); // td1[31:16] | tile_dim0
    g1[4] = (int)(tileRows & 0xffffu);                                 // tile_dim1 (tile_dim2=0)
    g1[5] = (int)(s0 & 0xffffffffu);                                   // dim0_stride[31:0]
    g1[6] = (int)((s0 >> 32) & 0xffffu);                               // dim0_stride[47:32]
    g1[7] = 0;
    i32x4 gz = {0, 0, 0, 0};
#if defined(__clang_major__) && (__clang_major__ >= 23)
    i32x8 gz8 = {0, 0, 0, 0, 0, 0, 0, 0};
    __builtin_amdgcn_tensor_load_to_lds(g0, g1, gz, gz, gz8, 0);
#else
    __builtin_amdgcn_tensor_load_to_lds(g0, g1, gz, gz, 0);
#endif
}
#define STAGE(DST, GS, STRIDE) \
    do { if (threadIdx.x < 32) tdm_load_2d((DST), (GS), 256u, 32u, (unsigned long long)(STRIDE)); } while (0)
#define STAGE_WAIT(N) \
    do { if (threadIdx.x < 32) __builtin_amdgcn_s_wait_tensorcnt(N); } while (0)
#else
// Fallback: cooperative vector-load staging (all 256 threads)
__device__ __forceinline__ void coop_stage(_Float16* dst, const _Float16* gsrc, int strideElems) {
    // 32 rows x 256 cols f16 = 8192 halves; each thread copies 32 halves in 16B chunks
#pragma unroll
    for (int j = 0; j < 4; ++j) {
        int e = threadIdx.x * 8 + j * 2048;
        int r = e >> 8, c = e & 255;
        *(v8h*)(dst + e) = *(const v8h*)(gsrc + (size_t)r * strideElems + c);
    }
}
#define STAGE(DST, GS, STRIDE) coop_stage((DST), (GS), (int)(STRIDE))
#define STAGE_WAIT(N) do { } while (0)
#endif

// ---------------- Prep kernels ----------------
__global__ void prep_rows(const float* __restrict__ Q, const float* __restrict__ K,
                          _Float16* __restrict__ Qh, _Float16* __restrict__ Kh,
                          float* __restrict__ diagQ, float* __restrict__ diagK) {
    int row = blockIdx.x;
    const float* src = blockIdx.y ? K : Q;
    _Float16* dst = blockIdx.y ? Kh : Qh;
    float* dg = blockIdx.y ? diagK : diagQ;
    __shared__ float red[256];
    float acc = 0.f;
    size_t base = (size_t)row * Dc;
    for (int i = threadIdx.x; i < Dc; i += 256) {
        float x = src[base + i] * SCALEc;
        dst[base + i] = (_Float16)x;
        acc += x * x;
    }
    red[threadIdx.x] = acc;
    __syncthreads();
    for (int off = 128; off > 0; off >>= 1) {
        if (threadIdx.x < off) red[threadIdx.x] += red[threadIdx.x + off];
        __syncthreads();
    }
    if (threadIdx.x == 0) dg[row] = 0.5f * red[0];
}

__global__ void prep_proj(const float* __restrict__ proj, _Float16* __restrict__ projTh) {
    int i = blockIdx.x * 256 + threadIdx.x; // i = d*M + m
    int d = i >> 8, m = i & 255;
    projTh[i] = (_Float16)proj[(size_t)m * Dc + d];
}

__global__ void cvt_f16x4(const float* __restrict__ src, _Float16* __restrict__ dst) {
    size_t i = ((size_t)blockIdx.x * 256 + threadIdx.x) * 4;
#pragma unroll
    for (int j = 0; j < 4; ++j) dst[i + j] = (_Float16)src[i + j];
}

// ---------------- Fused dash GEMM ----------------
// grid (L/32, B, 2); 8 waves as 2(rows) x 4(cols); wave tile 16x64, block tile 32 x 256 (full M).
// z==0 (Q): fused rowmax+diag+exp epilogue -> Qf f16 directly (dashQ never hits memory).
// z==1 (K): stores dashK f32 + per-block max partial (global stab needs a cross-grid reduce).
__global__ void dash_gemm_fused(const _Float16* __restrict__ Qh, const _Float16* __restrict__ Kh,
                                const _Float16* __restrict__ projTh,
                                const float* __restrict__ diagQ,
                                _Float16* __restrict__ Qf, float* __restrict__ dashK,
                                float* __restrict__ bpart) {
    __shared__ __align__(32) _Float16 btile[2][32 * 256];
    __shared__ float redrow[2][4][16];
    __shared__ float stab[32];
    int b = blockIdx.y;
    int isK = blockIdx.z;
    const _Float16* Xh = isK ? Kh : Qh;
    int w = threadIdx.x >> 5;
    int wr = w >> 2, wc = w & 3;
    int lane = threadIdx.x & 31;
    int row0 = blockIdx.x * 32 + wr * 16;
    int n0 = wc * 64;
    const _Float16* Arow = Xh + (size_t)(b * Lc + row0) * Dc;
    v8f acc[4] = {{}, {}, {}, {}};

    STAGE(&btile[0][0], projTh, Mc);            // kk = 0 tile
    const int KSTEPS = Dc / 32;                 // 16
    for (int it = 0; it < KSTEPS; ++it) {
        int kk = it * 32;
        if (it + 1 < KSTEPS) { STAGE(&btile[(it + 1) & 1][0], projTh + (size_t)(kk + 32) * Mc, Mc); STAGE_WAIT(1); }
        else                 { STAGE_WAIT(0); }
        __syncthreads();
        const _Float16* bt = &btile[it & 1][0] + n0;
        v16h af = load_a_frag(Arow, Dc, kk);
#pragma unroll
        for (int t = 0; t < 4; ++t) {
            v16h bf = lds_b_frag(bt + 16 * t);
            acc[t] = WMMA_F16(af, bf, acc[t]);
        }
        __syncthreads();
    }

    // Wave-level row max over this wave's 64 columns (all-lane result per 16-lane half)
    float rmax[8];
#pragma unroll
    for (int i = 0; i < 8; ++i) {
        float v = fmaxf(fmaxf(acc[0][i], acc[1][i]), fmaxf(acc[2][i], acc[3][i]));
#pragma unroll
        for (int msk = 1; msk <= 8; msk <<= 1) v = fmaxf(v, __shfl_xor(v, msk, 32));
        rmax[i] = v;
    }
    if (lane == 0 || lane == 16) {
        int mb = (lane >> 4) << 3;
#pragma unroll
        for (int i = 0; i < 8; ++i) redrow[wr][wc][mb + i] = rmax[i];
    }
    __syncthreads();
    if (threadIdx.x < 32) {
        int r = threadIdx.x;
        float m = fmaxf(fmaxf(redrow[r >> 4][0][r & 15], redrow[r >> 4][1][r & 15]),
                        fmaxf(redrow[r >> 4][2][r & 15], redrow[r >> 4][3][r & 15]));
        stab[r] = m;
    }
    __syncthreads();

    int mb = (lane >> 4) << 3;
    int n = lane & 15;
    if (!isK) {
        _Float16* Crow = Qf + (size_t)(b * Lc + row0) * Mc + n0;
#pragma unroll
        for (int i = 0; i < 8; ++i) {
            float dgv = diagQ[b * Lc + row0 + mb + i];
            float sub = dgv + stab[wr * 16 + mb + i];
#pragma unroll
            for (int t = 0; t < 4; ++t) {
                float val = __expf(acc[t][i] - sub) * INV_SQRT_M + EPSc;
                Crow[(size_t)(mb + i) * Mc + 16 * t + n] = (_Float16)val;
            }
        }
    } else {
        float* Crow = dashK + (size_t)(b * Lc + row0) * Mc + n0;
#pragma unroll
        for (int t = 0; t < 4; ++t) store_c(Crow + 16 * t, Mc, acc[t]);
        if (threadIdx.x == 0) {
            float m = stab[0];
#pragma unroll
            for (int i = 1; i < 32; ++i) m = fmaxf(m, stab[i]);
            bpart[(b << 8) + blockIdx.x] = m; // 256 partials per batch (L/32)
        }
    }
}

__global__ void batchmax_k2(const float* __restrict__ part, float* __restrict__ bmax) {
    int b = blockIdx.x;
    __shared__ float red[256];
    red[threadIdx.x] = part[b * 256 + threadIdx.x];
    __syncthreads();
    for (int off = 128; off > 0; off >>= 1) {
        if (threadIdx.x < off) red[threadIdx.x] = fmaxf(red[threadIdx.x], red[threadIdx.x + off]);
        __syncthreads();
    }
    if (threadIdx.x == 0) bmax[b] = red[0];
}

// KfT (B,M,L) f16 (transposed so the KV GEMM's A-fragments are contiguous along L)
__global__ void featurize_k(const float* __restrict__ dashK, const float* __restrict__ diagK,
                            const float* __restrict__ bmax, _Float16* __restrict__ KfT) {
    size_t i0 = ((size_t)blockIdx.x * 256 + threadIdx.x) * 4;
    int b = (int)(i0 >> 21);
    int within = (int)(i0 & ((1u << 21) - 1));
    int l = within >> 8;
    int m0 = within & 255;
    float sub = diagK[b * Lc + l] + bmax[b];
#pragma unroll
    for (int j = 0; j < 4; ++j) {
        float v = __expf(dashK[i0 + j] - sub) * INV_SQRT_M + EPSc;
        KfT[((size_t)b * Mc + (m0 + j)) * Lc + l] = (_Float16)v;
    }
}

__global__ void ksum_kernel(const _Float16* __restrict__ KfT, float* __restrict__ Ksum) {
    int r = blockIdx.x; // b*M + m
    const _Float16* p = KfT + (size_t)r * Lc;
    __shared__ float red[256];
    float acc = 0.f;
    for (int i = threadIdx.x; i < Lc; i += 256) acc += (float)p[i];
    red[threadIdx.x] = acc;
    __syncthreads();
    for (int off = 128; off > 0; off >>= 1) {
        if (threadIdx.x < off) red[threadIdx.x] += red[threadIdx.x + off];
        __syncthreads();
    }
    if (threadIdx.x == 0) Ksum[r] = red[0];
}

// ---------------- GEMM 2: KV = KfT(MxL) @ Vh(LxD), K-dim = L, TDM-staged B ----------------
// grid (M/32, D/256, B)
__global__ void kv_gemm(const _Float16* __restrict__ KfT, const _Float16* __restrict__ Vh,
                        float* __restrict__ KV) {
    __shared__ __align__(32) _Float16 btile[2][32 * 256];
    int b = blockIdx.z;
    int w = threadIdx.x >> 5;
    int wr = w >> 2, wc = w & 3;
    int row0 = blockIdx.x * 32 + wr * 16;     // m
    int colb = blockIdx.y * 256;              // d block
    const _Float16* Arow = KfT + (size_t)(b * Mc + row0) * Lc;
    const _Float16* Bg = Vh + (size_t)b * Lc * Dc + colb;
    v8f acc[4] = {{}, {}, {}, {}};

    STAGE(&btile[0][0], Bg, Dc);
    const int KSTEPS = Lc / 32;               // 256
    for (int it = 0; it < KSTEPS; ++it) {
        int kk = it * 32;
        if (it + 1 < KSTEPS) { STAGE(&btile[(it + 1) & 1][0], Bg + (size_t)(kk + 32) * Dc, Dc); STAGE_WAIT(1); }
        else                 { STAGE_WAIT(0); }
        __syncthreads();
        const _Float16* bt = &btile[it & 1][0] + wc * 64;
        v16h af = load_a_frag(Arow, Lc, kk);
#pragma unroll
        for (int t = 0; t < 4; ++t) {
            v16h bf = lds_b_frag(bt + 16 * t);
            acc[t] = WMMA_F16(af, bf, acc[t]);
        }
        __syncthreads();
    }
    float* Crow = KV + (size_t)(b * Mc + row0) * Dc + colb + wc * 64;
#pragma unroll
    for (int t = 0; t < 4; ++t) store_c(Crow + 16 * t, Dc, acc[t]);
}

// Z[b,l] = 1 / (Qf[b,l,:].Ksum[b,:] + eps)
__global__ void z_kernel(const _Float16* __restrict__ Qf, const float* __restrict__ Ksum,
                         float* __restrict__ Z) {
    int row = blockIdx.x;
    int b = row >> 13;
    __shared__ float red[256];
    red[threadIdx.x] = (float)Qf[(size_t)row * Mc + threadIdx.x] * Ksum[b * Mc + threadIdx.x];
    __syncthreads();
    for (int off = 128; off > 0; off >>= 1) {
        if (threadIdx.x < off) red[threadIdx.x] += red[threadIdx.x + off];
        __syncthreads();
    }
    if (threadIdx.x == 0) Z[row] = 1.0f / (red[0] + EPSc);
}

// ---------------- GEMM 3: out = (Qf(LxM) @ KVh(MxD)) * Z, TDM-staged B ----------------
// grid (L/32, D/256, B)
__global__ void out_gemm(const _Float16* __restrict__ Qf, const _Float16* __restrict__ KVh,
                         const float* __restrict__ Z, float* __restrict__ out) {
    __shared__ __align__(32) _Float16 btile[2][32 * 256];
    int b = blockIdx.z;
    int w = threadIdx.x >> 5;
    int wr = w >> 2, wc = w & 3;
    int row0 = blockIdx.x * 32 + wr * 16;     // l
    int colb = blockIdx.y * 256;              // d block
    const _Float16* Arow = Qf + (size_t)(b * Lc + row0) * Mc;
    const _Float16* Bg = KVh + (size_t)b * Mc * Dc + colb;
    v8f acc[4] = {{}, {}, {}, {}};

    STAGE(&btile[0][0], Bg, Dc);
    const int KSTEPS = Mc / 32;               // 8
    for (int it = 0; it < KSTEPS; ++it) {
        int kk = it * 32;
        if (it + 1 < KSTEPS) { STAGE(&btile[(it + 1) & 1][0], Bg + (size_t)(kk + 32) * Dc, Dc); STAGE_WAIT(1); }
        else                 { STAGE_WAIT(0); }
        __syncthreads();
        const _Float16* bt = &btile[it & 1][0] + wc * 64;
        v16h af = load_a_frag(Arow, Mc, kk);
#pragma unroll
        for (int t = 0; t < 4; ++t) {
            v16h bf = lds_b_frag(bt + 16 * t);
            acc[t] = WMMA_F16(af, bf, acc[t]);
        }
        __syncthreads();
    }
    const float* Zrow = Z + (size_t)b * Lc + row0;
    int mb = (threadIdx.x >> 4) & 0 ; // (unused placeholder removed below)
    (void)mb;
    {
        int lane = threadIdx.x & 31;
        int n = lane & 15;
        int mbase = (lane >> 4) << 3;
        float* Crow = out + (size_t)(b * Lc + row0) * Dc + colb + wc * 64;
#pragma unroll
        for (int t = 0; t < 4; ++t)
#pragma unroll
            for (int i = 0; i < 8; ++i)
                Crow[(size_t)(mbase + i) * Dc + 16 * t + n] = acc[t][i] * Zrow[mbase + i];
    }
}

// ---------------- Launch ----------------
extern "C" void kernel_launch(void* const* d_in, const int* in_sizes, int n_in,
                              void* d_out, int out_size, void* d_ws, size_t ws_size,
                              hipStream_t stream) {
    (void)in_sizes; (void)n_in; (void)out_size; (void)ws_size;
    const float* Q    = (const float*)d_in[0];
    const float* K    = (const float*)d_in[1];
    const float* V    = (const float*)d_in[2];
    const float* proj = (const float*)d_in[3];
    float* out = (float*)d_out;

    char* ws = (char*)d_ws;
    size_t off = 0;
    auto alloc = [&](size_t bytes) -> char* {
        char* p = ws + off;
        off += (bytes + 255) & ~(size_t)255;
        return p;
    };
    const size_t BLD = (size_t)Bc * Lc * Dc;
    const size_t BLM = (size_t)Bc * Lc * Mc;
    const size_t BL  = (size_t)Bc * Lc;
    const size_t BMD = (size_t)Bc * Mc * Dc;

    _Float16* Qh     = (_Float16*)alloc(BLD * 2);
    _Float16* Kh     = (_Float16*)alloc(BLD * 2);
    _Float16* Vh     = (_Float16*)alloc(BLD * 2);
    _Float16* projTh = (_Float16*)alloc((size_t)Dc * Mc * 2);
    float*    dashK  = (float*)alloc(BLM * 4);
    _Float16* Qf     = (_Float16*)alloc(BLM * 2);
    _Float16* KfT    = (_Float16*)alloc(BLM * 2);
    float*    diagQ  = (float*)alloc(BL * 4);
    float*    diagK  = (float*)alloc(BL * 4);
    float*    bpart  = (float*)alloc((size_t)Bc * 256 * 4);
    float*    bmax   = (float*)alloc((size_t)Bc * 4);
    float*    Ksum   = (float*)alloc((size_t)Bc * Mc * 4);
    float*    Zbuf   = (float*)alloc(BL * 4);
    float*    KV     = (float*)alloc(BMD * 4);
    _Float16* KVh    = (_Float16*)alloc(BMD * 2);

    // 1) prep
    prep_rows<<<dim3(Bc * Lc, 2), 256, 0, stream>>>(Q, K, Qh, Kh, diagQ, diagK);
    prep_proj<<<(Dc * Mc) / 256, 256, 0, stream>>>(proj, projTh);
    cvt_f16x4<<<(unsigned)(BLD / 1024), 256, 0, stream>>>(V, Vh);

    // 2) dash GEMMs: Q fully fused -> Qf; K -> dashK + per-block max
    dash_gemm_fused<<<dim3(Lc / 32, Bc, 2), 256, 0, stream>>>(Qh, Kh, projTh, diagQ,
                                                              Qf, dashK, bpart);

    // 3) K stabilizer + featurize
    batchmax_k2<<<Bc, 256, 0, stream>>>(bpart, bmax);
    featurize_k<<<(unsigned)(BLM / 1024), 256, 0, stream>>>(dashK, diagK, bmax, KfT);

    // 4) Ksum, KV GEMM, Z
    ksum_kernel<<<Bc * Mc, 256, 0, stream>>>(KfT, Ksum);
    kv_gemm<<<dim3(Mc / 32, Dc / 256, Bc), 256, 0, stream>>>(KfT, Vh, KV);
    z_kernel<<<Bc * Lc, 256, 0, stream>>>(Qf, Ksum, Zbuf);
    cvt_f16x4<<<(unsigned)(BMD / 1024), 256, 0, stream>>>(KV, KVh);

    // 5) output GEMM with Z scaling
    out_gemm<<<dim3(Lc / 32, Dc / 256, Bc), 256, 0, stream>>>(Qf, KVh, Zbuf, out);
}
